// MeshConv_80607946211848
// MI455X (gfx1250) — matrix operand order repaired and checked by reading, not compiled, not run on hardware
//
#include <hip/hip_runtime.h>

typedef __attribute__((ext_vector_type(16))) _Float16 v16h;
typedef __attribute__((ext_vector_type(8)))  _Float16 v8h;
typedef __attribute__((ext_vector_type(8)))  float    v8f;
typedef __attribute__((ext_vector_type(4)))  float    v4f;
typedef __attribute__((ext_vector_type(4)))  int      v4i;

#define BATCH   4
#define C_IN    32
#define C_OUT   64
#define EDGES   200000
#define KW      5
#define TILE_E  16
#define TILES_PER_B (EDGES / TILE_E)        // 12500
#define WAVES_PER_BLOCK 8
#define NFRAG_HALFS (KW * 4 * 32 * 16)      // W fragments: 10240 halfs = 20 KB
#define STAGE_ITERS (NFRAG_HALFS / 256)     // 40 staging elements per thread

// One wave computes out[b, 0:64, e0:e0+16] via v_wmma_f32_16x16x32_f16.
//   M = 16 edges, N = 16 outputs (x4 tiles), K = 160 = 5 groups x 32 channels.
__global__ __launch_bounds__(256) void meshconv_wmma_kernel(
    const float* __restrict__ x,    // (B, C_IN, E)   -- L2-resident, RT loads
    const int*   __restrict__ Gi,   // (B, E, 5)      -- streamed once, NT loads
    const float* __restrict__ W,    // (C_OUT, C_IN, 1, KW)
    const float* __restrict__ bias, // (C_OUT)
    float* __restrict__ out)        // (B, C_OUT, E, 1) -- streamed once, NT stores
{
    // W staged in B-fragment layout: lane l holds column n=(l&15),
    // K(channel) = (l>>4)*16 + i for i=0..15 (16 contiguous halfs per lane).
    __shared__ _Float16 wfrag[NFRAG_HALFS];                       // 20 KB
    __shared__ _Float16 gtile[WAVES_PER_BLOCK][TILE_E][KW][C_IN]; // 40 KB
    __shared__ int      gidx[WAVES_PER_BLOCK][TILE_E * KW];       // 2.5 KB

    const int tid  = threadIdx.x;
    const int lane = tid & 31;
    const int wid  = tid >> 5;

    // ---- stage W fragments (whole block, once) ----
    // Fully unrolled: the 40 independent loads per thread get claused and
    // covered by incremental s_wait_loadcnt instead of 40 serialized waits.
    {
        float wv[STAGE_ITERS];
        #pragma unroll
        for (int it = 0; it < STAGE_ITERS; ++it) {
            const int u = tid + it * 256;
            const int i = u & 15;            // half index within lane
            const int l = (u >> 4) & 31;     // lane
            const int t = (u >> 9) & 3;      // N tile
            const int j = u >> 11;           // K group (kernel tap)
            const int n = t * 16 + (l & 15); // output channel
            const int c = (l >> 4) * 16 + i; // input channel
            wv[it] = W[(n * C_IN + c) * KW + j];
        }
        #pragma unroll
        for (int it = 0; it < STAGE_ITERS; ++it) {
            wfrag[tid + it * 256] = (_Float16)wv[it];
        }
    }
    __syncthreads();

    const int gwave = blockIdx.x * WAVES_PER_BLOCK + wid;
    const int b     = gwave / TILES_PER_B;
    const int tile  = gwave % TILES_PER_B;
    const int e0    = tile * TILE_E;

    const float* xb = x + (size_t)b * C_IN * EDGES;
    const int    c  = lane;  // lane == input channel during gather

    // ---- coalesced fetch of the 16x5 index block (320 contiguous bytes) ----
    // lanes 0..19 each load one b128 (NT: Gi is read exactly once).
    {
        const int* ib = Gi + ((size_t)b * EDGES + e0) * KW;
        if (lane < (TILE_E * KW) / 4) {
            const v4i v = __builtin_nontemporal_load((const v4i*)ib + lane);
            *(v4i*)&gidx[wid][lane * 4] = v;
        }
    }
    // Same-wave LDS ops are in-order; no cross-wave sharing of gidx/gtile.

    // ---- gather one-ring neighbors, build symmetric features in f16 ----
    #pragma unroll 8
    for (int m = 0; m < TILE_E; ++m) {
        const int* gp = &gidx[wid][m * KW];
        const int g0 = gp[0], g1 = gp[1], g2 = gp[2], g3 = gp[3], g4 = gp[4];
        const float f0 = xb[(size_t)c * EDGES + g0];
        const float f1 = xb[(size_t)c * EDGES + g1];
        const float f2 = xb[(size_t)c * EDGES + g2];
        const float f3 = xb[(size_t)c * EDGES + g3];
        const float f4 = xb[(size_t)c * EDGES + g4];
        gtile[wid][m][0][c] = (_Float16)f0;
        gtile[wid][m][1][c] = (_Float16)(f1 + f3);
        gtile[wid][m][2][c] = (_Float16)(f2 + f4);
        gtile[wid][m][3][c] = (_Float16)fabsf(f1 - f3);
        gtile[wid][m][4][c] = (_Float16)fabsf(f2 - f4);
    }
    // Same-wave LDS RAW: make staged data visible before ds fragment reads.
    asm volatile("s_wait_dscnt 0x0" ::: "memory");

    // ---- accumulators seeded with bias (C/D layout: lane holds n=(lane&15)) ----
    v8f acc[4];
    #pragma unroll
    for (int t = 0; t < 4; ++t) {
        const float bv = bias[t * 16 + (lane & 15)];
        #pragma unroll
        for (int r = 0; r < 8; ++r) acc[t][r] = bv;
    }

    // ---- 5 K-steps x 4 N-tiles of v_wmma_f32_16x16x32_f16 ----
    const int m_row = lane & 15;
    const int hi    = lane >> 4;
    #pragma unroll
    for (int j = 0; j < KW; ++j) {
        // A-fragment in contiguous K order (matches wfrag):
        // lane<16: K 0..15 ; lane>=16: K 16..31
        const _Float16* arow = &gtile[wid][m_row][j][0];
        const v8h alo = *(const v8h*)(arow + hi * 16);
        const v8h ahi = *(const v8h*)(arow + 8 + hi * 16);
        v16h a;
        #pragma unroll
        for (int i = 0; i < 8; ++i) { a[i] = alo[i]; a[i + 8] = ahi[i]; }

        #pragma unroll
        for (int t = 0; t < 4; ++t) {
            const v16h bf = *(const v16h*)&wfrag[((j * 4 + t) * 32 + lane) * 16];
            acc[t] = __builtin_amdgcn_wmma_f32_16x16x32_f16(
                false, a, false, bf, (short)0, acc[t], false, false);
        }
    }

    // ---- store: D layout gives each lane 8 consecutive edges per N-tile ----
    // NT stores: output is write-once; keep L2 capacity for the x gather set.
    float* ob = out + (size_t)b * C_OUT * EDGES;
    const int n_lo = lane & 15;
    const int moff = hi * 8;
    #pragma unroll
    for (int t = 0; t < 4; ++t) {
        const int o = t * 16 + n_lo;
        float* p = ob + (size_t)o * EDGES + e0 + moff;
        v4f lo  = {acc[t][0], acc[t][1], acc[t][2], acc[t][3]};
        v4f hi4 = {acc[t][4], acc[t][5], acc[t][6], acc[t][7]};
        __builtin_nontemporal_store(lo,  (v4f*)(p));
        __builtin_nontemporal_store(hi4, (v4f*)(p + 4));
    }
}

extern "C" void kernel_launch(void* const* d_in, const int* in_sizes, int n_in,
                              void* d_out, int out_size, void* d_ws, size_t ws_size,
                              hipStream_t stream) {
    const float* x    = (const float*)d_in[0];
    const int*   Gi   = (const int*)d_in[1];
    const float* W    = (const float*)d_in[2];
    const float* bias = (const float*)d_in[3];
    float* out        = (float*)d_out;

    const int total_waves = BATCH * TILES_PER_B;          // 50000
    const int blocks = total_waves / WAVES_PER_BLOCK;     // 6250
    meshconv_wmma_kernel<<<blocks, WAVES_PER_BLOCK * 32, 0, stream>>>(
        x, Gi, W, bias, out);
}
